// GraphAttentionV2_10642928959905
// MI455X (gfx1250) — matrix operand order repaired and checked by reading, not compiled, not run on hardware
//
#include <hip/hip_runtime.h>
#include <cstdint>
#include <cstddef>

typedef __attribute__((ext_vector_type(16))) _Float16 v16h;
typedef __attribute__((ext_vector_type(8)))  _Float16 v8h;
typedef __attribute__((ext_vector_type(8)))  float    v8f;

#define B_      16
#define N_      48
#define NN      768       // B_*N_
#define FN      5
#define H1_     32
#define H2_     64
#define EPB     4512      // 2*N*(N-1) edges per batch
#define E_      72192     // B_*EPB
#define S_      2256      // kept edges per batch
#define EK      36096     // B_*S_
#define SPAD    2272      // S_ padded to multiple of 32
#define D_      192
#define QT      141       // S_/16

// ---------------- WMMA fragment helpers (CDNA5 wave32 layouts, 32-bit offsets) --------

__device__ __forceinline__ v16h load_a16(const _Float16* __restrict__ base, int ld) {
  // A 16x32 f16: lane = m + 16*h ; halfs[0..7]=A[m][8h+0..7], halfs[8..15]=A[m][16+8h+0..7]
  int lane = threadIdx.x & 31;
  int m = lane & 15, h = lane >> 4;
  int o = m * ld + h * 8;
  v8h lo = *(const v8h*)(base + o);
  v8h hi = *(const v8h*)(base + o + 16);
  v16h r;
#pragma unroll
  for (int i = 0; i < 8; ++i) { r[i] = lo[i]; r[8 + i] = hi[i]; }
  return r;
}

__device__ __forceinline__ v16h load_b16(const _Float16* __restrict__ base, int ld) {
  // B 32x16 f16: lane = n + 16*h ; halfs[0..7]=B[8h+0..7][n], halfs[8..15]=B[16+8h+0..7][n]
  int lane = threadIdx.x & 31;
  int n = lane & 15, h = lane >> 4;
  v16h r;
#pragma unroll
  for (int i = 0; i < 8; ++i) { r[i] = base[(8 * h + i) * ld + n]; r[8 + i] = base[(16 + 8 * h + i) * ld + n]; }
  return r;
}

// ---------------- small prologue kernels ----------------

__global__ void k_zero(uint32_t* __restrict__ p, size_t nwords) {
  size_t t = (size_t)blockIdx.x * blockDim.x + threadIdx.x;
  size_t stride = (size_t)gridDim.x * blockDim.x;
  for (; t < nwords; t += stride) p[t] = 0u;
}

__global__ void k_edgew(const float* __restrict__ ea, const float* __restrict__ Wwe,
                        const float* __restrict__ bwe, float* __restrict__ w) {
  int e = blockIdx.x * blockDim.x + threadIdx.x;
  if (e >= E_) return;
  float wr  = ea[2 * e];
  float cls = ea[2 * e + 1];
  float c0 = (cls == 0.f) ? 1.f : 0.f;
  float v = Wwe[0] * wr + Wwe[1] * c0 + Wwe[2] * (1.f - c0) + bwe[0];
  w[e] = fmaxf(v, 0.f);
}

// deterministic gather-based GraphConv aggregation (no atomics)
template <int F>
__global__ void k_agg(const float* __restrict__ xin, const float* __restrict__ w,
                      float* __restrict__ agg) {
  int t = blockIdx.x * blockDim.x + threadIdx.x;
  if (t >= NN * F) return;
  int nd = t / F, f = t - nd * F;
  int b = nd / N_, i = nd - b * N_;
  float acc = 0.f;
  for (int s = 0; s < N_; ++s) {
    if (s == i) continue;
    int p = s * (N_ - 1) + (i < s ? i : i - 1);   // (src=s, dst=i) pair index
    int e = b * EPB + 2 * p;                       // class-0 edge; class-1 = e+1
    float xv = xin[(b * N_ + s) * F + f];
    acc += xv * (w[e] + w[e + 1]);
  }
  agg[t] = acc;
}

template <int FI, int FO>
__global__ void k_lin(const float* __restrict__ agg, const float* __restrict__ xin,
                      const float* __restrict__ Wrel, const float* __restrict__ brel,
                      const float* __restrict__ Wroot, float* __restrict__ xout) {
  int t = blockIdx.x * blockDim.x + threadIdx.x;
  if (t >= NN * FO) return;
  int nd = t / FO, o = t - nd * FO;
  float acc = brel[o];
#pragma unroll 4
  for (int f = 0; f < FI; ++f)
    acc += agg[nd * FI + f] * Wrel[o * FI + f] + xin[nd * FI + f] * Wroot[o * FI + f];
  xout[t] = fmaxf(acc, 0.f);
}

// fuse the two chained projections: Weff = W_in_part @ W_qkv_part  (stored transposed f16)
__global__ void k_fusew(const float* __restrict__ Win, const float* __restrict__ Wqkv,
                        _Float16* __restrict__ Wt) {
  int t = blockIdx.x * blockDim.x + threadIdx.x;
  if (t >= 3 * D_ * D_) return;
  int part = t / (D_ * D_);
  int rem = t - part * D_ * D_;
  int n = rem / D_, k = rem - n * D_;
  float acc = 0.f;
  for (int j = 0; j < D_; ++j)
    acc += Win[(part * D_ + n) * D_ + j] * Wqkv[(part * D_ + j) * D_ + k];
  Wt[part * D_ * D_ + k * D_ + n] = (_Float16)acc;  // transposed: [in k][out n]
}

__global__ void k_fuseb(const float* __restrict__ Win, const float* __restrict__ bqkv,
                        const float* __restrict__ bin, float* __restrict__ beff) {
  int t = blockIdx.x * blockDim.x + threadIdx.x;
  if (t >= 3 * D_) return;
  int part = t / D_, n = t - part * D_;
  float acc = bin[t];
  for (int j = 0; j < D_; ++j)
    acc += Win[(part * D_ + n) * D_ + j] * bqkv[part * D_ + j];
  beff[t] = acc;
}

// fuse W_out then W_o into one 192-vector + scalar bias (wo[0..191], wo[192]=bias)
__global__ void k_fuseo(const float* __restrict__ Wo, const float* __restrict__ Wout,
                        const float* __restrict__ bout, const float* __restrict__ bo,
                        float* __restrict__ wo) {
  int t = blockIdx.x * blockDim.x + threadIdx.x;
  if (t < D_) {
    float acc = 0.f;
    for (int j = 0; j < D_; ++j) acc += Wo[j] * Wout[j * D_ + t];
    wo[t] = acc;
  }
  if (t == 0) {
    float acc = bo[0];
    for (int j = 0; j < D_; ++j) acc += Wo[j] * bout[j];
    wo[D_] = acc;
  }
}

// build feat rows for kept (src>dst) edges; kept index computed analytically
__global__ void k_build(const float* __restrict__ ea, const float* __restrict__ x2,
                        const float* __restrict__ Wee, const float* __restrict__ bee,
                        _Float16* __restrict__ feat, int* __restrict__ esrc,
                        int* __restrict__ edst, float* __restrict__ clsK) {
  int e = blockIdx.x;
  int b = e / EPB;
  int r0 = e - b * EPB;
  int p = r0 >> 1, c = r0 & 1;
  int i = p / (N_ - 1);
  int jt = p - i * (N_ - 1);
  int j = jt + (jt >= i ? 1 : 0);
  if (i <= j) return;                     // keep only src > dst
  int r = (i * (i - 1)) / 2 + j;          // rank among kept pairs within batch
  int kk = b * S_ + 2 * r + c;
  int t = threadIdx.x;                    // 0..63
  int sn = b * N_ + i, dn = b * N_ + j;
  float wr = ea[2 * e];
  float c0 = (c == 0) ? 1.f : 0.f;
  float ee = Wee[t * 3 + 0] * wr + Wee[t * 3 + 1] * c0 + Wee[t * 3 + 2] * (1.f - c0) + bee[t];
  ee = fmaxf(ee, 0.f);
  int base = kk * D_;
  feat[base + t]        = (_Float16)x2[sn * H2_ + t];
  feat[base + 64 + t]   = (_Float16)ee;
  feat[base + 128 + t]  = (_Float16)x2[dn * H2_ + t];
  if (t == 0) { esrc[kk] = sn; edst[kk] = dn; clsK[kk] = (float)c; }
}

// ---------------- fused QKV projection GEMM (WMMA f16, f32 acc) ----------------
// one wave computes a 16-row tile across all 3 outputs x 12 N-tiles (216 WMMAs, A reused)
__global__ __launch_bounds__(32) void k_qkv(const _Float16* __restrict__ feat,
                                            const _Float16* __restrict__ Wt,
                                            const float* __restrict__ beff,
                                            _Float16* __restrict__ Qh,
                                            _Float16* __restrict__ Kt,
                                            _Float16* __restrict__ Vh) {
  __shared__ __align__(16) _Float16 st[16 * 16];   // store-bounce tile
  const int blk = blockIdx.x;
  const int b  = blk / QT;
  const int s0 = (blk - b * QT) * 16;
  const int m0 = b * S_ + s0;
  const int lane = threadIdx.x & 31;
  const int col = lane & 15, h = lane >> 4;

  v16h a[6];
#pragma unroll
  for (int j = 0; j < 6; ++j)
    a[j] = load_a16(feat + m0 * D_ + j * 32, D_);

  for (int part = 0; part < 3; ++part) {
    const _Float16* W = Wt + part * D_ * D_;
#pragma unroll
    for (int n = 0; n < 12; ++n) {
      float bv = beff[part * D_ + n * 16 + col];
      v8f cacc;
#pragma unroll
      for (int r = 0; r < 8; ++r) cacc[r] = bv;
      // software-pipelined B fragments: next load in flight during current WMMA
      v16h bb = load_b16(W + n * 16, D_);
#pragma unroll
      for (int j = 0; j < 6; ++j) {
        v16h nb;
        if (j < 5) nb = load_b16(W + (j + 1) * 32 * D_ + n * 16, D_);
        cacc = __builtin_amdgcn_wmma_f32_16x16x32_f16(false, a[j], false, bb,
                                                      (short)0, cacc, false, false);
        bb = nb;
      }
      if (part == 1) {
        // K^T: rows r are contiguous in token dim -> one packed b128 store per lane
        v8h pk;
#pragma unroll
        for (int r = 0; r < 8; ++r) pk[r] = (_Float16)cacc[r];
        *(v8h*)(Kt + (b * D_ + n * 16 + col) * SPAD + s0 + 8 * h) = pk;
      } else {
        // Q/V row-major: bounce via LDS so each lane does one contiguous b128 store
#pragma unroll
        for (int r = 0; r < 8; ++r) st[(r + 8 * h) * 16 + col] = (_Float16)cacc[r];
        __syncthreads();
        int rr = lane >> 1, cc = (lane & 1) * 8;
        v8h pk = *(const v8h*)(st + rr * 16 + cc);
        _Float16* dst = (part == 0)
            ? (Qh + (m0 + rr) * D_ + n * 16 + cc)
            : (Vh + (b * SPAD + s0 + rr) * D_ + n * 16 + cc);
        *(v8h*)dst = pk;
        __syncthreads();
      }
    }
  }
}

// ---------------- flash attention + fused output projection ----------------
__global__ __launch_bounds__(32) void k_flash(const _Float16* __restrict__ Qh,
                                              const _Float16* __restrict__ Kt,
                                              const _Float16* __restrict__ Vh,
                                              const float* __restrict__ wob,
                                              float* __restrict__ logit) {
  __shared__ __align__(16) _Float16 plds[16 * 32];
  const int b = blockIdx.y, qt = blockIdx.x;
  const int lane = threadIdx.x & 31;
  const int col = lane & 15, h = lane >> 4;
  const int q0 = b * S_ + qt * 16;
  const float scal = 0.07216878364870323f;  // 1/sqrt(192)
  const _Float16* Kb = Kt + b * D_ * SPAD;  // [d][token], ld = SPAD
  const _Float16* Vb = Vh + b * SPAD * D_;  // [token][d], ld = D_

  v16h qa[6];
#pragma unroll
  for (int j = 0; j < 6; ++j)
    qa[j] = load_a16(Qh + q0 * D_ + j * 32, D_);

  v8f O[12];
  v8f vz = {};
#pragma unroll
  for (int n = 0; n < 12; ++n) O[n] = vz;
  float mrow[8], lrow[8];
#pragma unroll
  for (int r = 0; r < 8; ++r) { mrow[r] = -1e30f; lrow[r] = 0.f; }

  for (int kv = 0; kv < SPAD; kv += 32) {
    if (kv + 32 < SPAD) {
      __builtin_prefetch(Kb + kv + 32, 0, 1);
      __builtin_prefetch(Vb + (kv + 32) * D_, 0, 1);
    }
    // scores: two 16x16 tiles (32 keys), K-dim d=192 in 6 steps, double-buffered B
    v8f c1 = vz, c2 = vz;
    v16h b1 = load_b16(Kb + kv, SPAD);
    v16h b2 = load_b16(Kb + kv + 16, SPAD);
#pragma unroll
    for (int j = 0; j < 6; ++j) {
      v16h n1, n2;
      if (j < 5) {
        n1 = load_b16(Kb + (j + 1) * 32 * SPAD + kv, SPAD);
        n2 = load_b16(Kb + (j + 1) * 32 * SPAD + kv + 16, SPAD);
      }
      c1 = __builtin_amdgcn_wmma_f32_16x16x32_f16(false, qa[j], false, b1, (short)0, c1, false, false);
      c2 = __builtin_amdgcn_wmma_f32_16x16x32_f16(false, qa[j], false, b2, (short)0, c2, false, false);
      b1 = n1; b2 = n2;
    }
    const bool oob2 = (kv + 16 + col) >= S_;
    float scl[8];
#pragma unroll
    for (int r = 0; r < 8; ++r) {
      float s1 = c1[r] * scal;
      float s2 = oob2 ? -1e30f : c2[r] * scal;
      float t = fmaxf(s1, s2);
#pragma unroll
      for (int off = 8; off >= 1; off >>= 1) t = fmaxf(t, __shfl_xor(t, off, 32));
      float mn  = fmaxf(mrow[r], t);
      float esc = __expf(mrow[r] - mn);
      float p1  = __expf(s1 - mn);
      float p2  = __expf(s2 - mn);
      float rs  = p1 + p2;
#pragma unroll
      for (int off = 8; off >= 1; off >>= 1) rs += __shfl_xor(rs, off, 32);
      lrow[r] = lrow[r] * esc + rs;
      mrow[r] = mn;
      scl[r]  = esc;
      plds[(r + 8 * h) * 32 + col]      = (_Float16)p1;
      plds[(r + 8 * h) * 32 + 16 + col] = (_Float16)p2;
    }
#pragma unroll
    for (int n = 0; n < 12; ++n)
#pragma unroll
      for (int r = 0; r < 8; ++r) O[n][r] *= scl[r];
    __syncthreads();
    // P in A-fragment layout from LDS
    v8h lo = *(const v8h*)(plds + col * 32 + 8 * h);
    v8h hi = *(const v8h*)(plds + col * 32 + 16 + 8 * h);
    v16h pa;
#pragma unroll
    for (int i = 0; i < 8; ++i) { pa[i] = lo[i]; pa[8 + i] = hi[i]; }
    // O += P @ V (K=32 tokens), double-buffered B
    v16h vb = load_b16(Vb + kv * D_, D_);
#pragma unroll
    for (int n = 0; n < 12; ++n) {
      v16h nb;
      if (n < 11) nb = load_b16(Vb + kv * D_ + (n + 1) * 16, D_);
      O[n] = __builtin_amdgcn_wmma_f32_16x16x32_f16(false, pa, false, vb, (short)0, O[n], false, false);
      vb = nb;
    }
    __syncthreads();
  }

  // fused (W_o @ W_out) projection: logit = (O/l) . wo + bias
  float acc[8];
#pragma unroll
  for (int r = 0; r < 8; ++r) acc[r] = 0.f;
#pragma unroll
  for (int n = 0; n < 12; ++n) {
    float wv = wob[n * 16 + col];
#pragma unroll
    for (int r = 0; r < 8; ++r) acc[r] += O[n][r] * wv;
  }
#pragma unroll
  for (int r = 0; r < 8; ++r) {
    float v = acc[r] / lrow[r];
#pragma unroll
    for (int off = 8; off >= 1; off >>= 1) v += __shfl_xor(v, off, 32);
    if (col == 0) logit[q0 + r + 8 * h] = v + wob[D_];
  }
}

// ---------------- final pairwise argmin selection ----------------
__global__ void k_out(const float* __restrict__ logit, const int* __restrict__ esrc,
                      const int* __restrict__ edst, const float* __restrict__ clsK,
                      float* __restrict__ out) {
  int t = blockIdx.x * blockDim.x + threadIdx.x;
  if (t >= B_ * (S_ / 2)) return;
  int b = t / (S_ / 2), i = t - b * (S_ / 2);
  int base = b * S_ + 2 * i;
  float l0 = logit[base], l1 = logit[base + 1];
  int mi = (l1 < l0) ? 1 : 0;   // argmin, first-min wins ties
  int sel = base + mi;
  out[t * 2]     = (float)esrc[sel];
  out[t * 2 + 1] = (float)edst[sel];
  out[B_ * S_ + t]                 = fminf(l0, l1);
  out[B_ * S_ + B_ * (S_ / 2) + t] = clsK[sel];
}

// ---------------- launch ----------------
extern "C" void kernel_launch(void* const* d_in, const int* in_sizes, int n_in,
                              void* d_out, int out_size, void* d_ws, size_t ws_size,
                              hipStream_t stream) {
  const float* x     = (const float*)d_in[0];
  const float* ea    = (const float*)d_in[2];
  const float* Wwe   = (const float*)d_in[5];
  const float* bwe   = (const float*)d_in[6];
  const float* Wrel1 = (const float*)d_in[7];
  const float* brel1 = (const float*)d_in[8];
  const float* Wroot1= (const float*)d_in[9];
  const float* Wrel2 = (const float*)d_in[10];
  const float* brel2 = (const float*)d_in[11];
  const float* Wroot2= (const float*)d_in[12];
  const float* Wee   = (const float*)d_in[13];
  const float* bee   = (const float*)d_in[14];
  const float* Wqkv  = (const float*)d_in[15];
  const float* bqkv  = (const float*)d_in[16];
  const float* Win   = (const float*)d_in[17];
  const float* bin   = (const float*)d_in[18];
  const float* Wout  = (const float*)d_in[19];
  const float* bout  = (const float*)d_in[20];
  const float* Wo    = (const float*)d_in[21];
  const float* bo    = (const float*)d_in[22];

  char* ws = (char*)d_ws;
  size_t off = 0;
  auto take = [&](size_t bytes) -> char* {
    char* p = ws + off;
    off = (off + bytes + 255) & ~(size_t)255;
    return p;
  };
  float*     w_edge = (float*)take((size_t)E_ * 4);
  float*     agg1   = (float*)take((size_t)NN * FN * 4);
  float*     x1     = (float*)take((size_t)NN * H1_ * 4);
  float*     agg2   = (float*)take((size_t)NN * H1_ * 4);
  float*     x2     = (float*)take((size_t)NN * H2_ * 4);
  _Float16*  feat   = (_Float16*)take((size_t)EK * D_ * 2);
  _Float16*  Qh     = (_Float16*)take((size_t)EK * D_ * 2);
  _Float16*  Kt     = (_Float16*)take((size_t)B_ * D_ * SPAD * 2);
  _Float16*  Vh     = (_Float16*)take((size_t)B_ * SPAD * D_ * 2);
  _Float16*  Wt     = (_Float16*)take((size_t)3 * D_ * D_ * 2);
  float*     beff   = (float*)take((size_t)3 * D_ * 4);
  float*     wob    = (float*)take((size_t)(D_ + 1) * 4);
  float*     logit  = (float*)take((size_t)EK * 4);
  int*       esrc   = (int*)take((size_t)EK * 4);
  int*       edst   = (int*)take((size_t)EK * 4);
  float*     clsK   = (float*)take((size_t)EK * 4);

  // zero padded K/V buffers (pad tokens must read as zero / masked)
  size_t kw = (size_t)B_ * D_ * SPAD / 2;  // uint32 words
  k_zero<<<2048, 256, 0, stream>>>((uint32_t*)Kt, kw);
  k_zero<<<2048, 256, 0, stream>>>((uint32_t*)Vh, kw);

  k_edgew<<<(E_ + 255) / 256, 256, 0, stream>>>(ea, Wwe, bwe, w_edge);

  k_agg<FN><<<(NN * FN + 127) / 128, 128, 0, stream>>>(x, w_edge, agg1);
  k_lin<FN, H1_><<<(NN * H1_ + 255) / 256, 256, 0, stream>>>(agg1, x, Wrel1, brel1, Wroot1, x1);
  k_agg<H1_><<<(NN * H1_ + 127) / 128, 128, 0, stream>>>(x1, w_edge, agg2);
  k_lin<H1_, H2_><<<(NN * H2_ + 255) / 256, 256, 0, stream>>>(agg2, x1, Wrel2, brel2, Wroot2, x2);

  k_fusew<<<(3 * D_ * D_ + 255) / 256, 256, 0, stream>>>(Win, Wqkv, Wt);
  k_fuseb<<<(3 * D_ + 63) / 64, 64, 0, stream>>>(Win, bqkv, bin, beff);
  k_fuseo<<<1, 256, 0, stream>>>(Wo, Wout, bout, bo, wob);

  k_build<<<E_, 64, 0, stream>>>(ea, x2, Wee, bee, feat, esrc, edst, clsK);

  k_qkv<<<B_ * QT, 32, 0, stream>>>(feat, Wt, beff, Qh, Kt, Vh);

  k_flash<<<dim3(QT, B_), 32, 0, stream>>>(Qh, Kt, Vh, wob, logit);

  k_out<<<(B_ * (S_ / 2) + 255) / 256, 256, 0, stream>>>(logit, esrc, edst, clsK, (float*)d_out);
}
